// BertAttention_82205674046030
// MI455X (gfx1250) — matrix operand order repaired and checked by reading, not compiled, not run on hardware
//
#include <hip/hip_runtime.h>

// BERT attention layer for MI455X (gfx1250, wave32, WMMA).
// B=8, S=1024, H=1024, NH=16, HD=64. All I/O fp32; internals bf16 WMMA.
// Projection GEMMs stage weight tiles in LDS via async global->LDS copies
// (ASYNCcnt path), double-buffered; attention is flash-style with online
// softmax and a per-wave LDS bounce to re-swizzle probs C-layout -> A-layout.

#define B_  8
#define S_  1024
#define H_  1024
#define NH_ 16
#define HD_ 64
#define M_  (B_ * S_)   // 8192 tokens

typedef __attribute__((ext_vector_type(16))) __bf16 v16bf;
typedef __attribute__((ext_vector_type(8)))  float  v8f;
typedef __attribute__((ext_vector_type(4)))  unsigned int u32x4;
typedef __attribute__((address_space(3)))    __bf16 bf16_lds;

union BF16Frag { v16bf v; u32x4 q[2]; };

// B-operand fragment: 16 contiguous bf16 (lane supplies K-run 16*(lane>>4)+j).
__device__ __forceinline__ v16bf ldfragB(const __bf16* p) {
  BF16Frag u;
  u.q[0] = *(const u32x4*)p;
  u.q[1] = *(const u32x4*)(p + 8);
  return u.v;
}
// A-operand fragment: 16 contiguous bf16 [base..base+15]
// (K = j + (lane>>4)*16 relative to the 32-wide K chunk).
__device__ __forceinline__ v16bf ldfragA(const __bf16* p) {
  BF16Frag u;
  u.q[0] = *(const u32x4*)p;
  u.q[1] = *(const u32x4*)(p + 8);
  return u.v;
}
__device__ __forceinline__ v8f wmma_bf16(v16bf a, v16bf b, v8f c) {
  return __builtin_amdgcn_wmma_f32_16x16x32_bf16(
      /*neg_a=*/false, a, /*neg_b=*/false, b,
      /*c_mod=*/(short)0, c, /*reuse_a=*/false, /*reuse_b=*/false);
}

// 16-byte async copy global -> LDS (tracked by ASYNCcnt, not LOADcnt).
__device__ __forceinline__ void async_copy16(__bf16* lds_dst,
                                             const __bf16* gsrc) {
  unsigned lds_a = (unsigned)(size_t)(bf16_lds*)lds_dst;  // 32-bit LDS addr
  unsigned long long ga = (unsigned long long)(size_t)gsrc;
  asm volatile("global_load_async_to_lds_b128 %0, %1, off"
               :: "v"(lds_a), "v"(ga) : "memory");
}
__device__ __forceinline__ void wait_async0() {
  asm volatile("s_wait_asynccnt 0" ::: "memory");
}

// ---------------------------------------------------------------- convert
__global__ void cvt_f32_bf16(const float* __restrict__ src,
                             __bf16* __restrict__ dst, int n) {
  int i = blockIdx.x * blockDim.x + threadIdx.x;
  if (i < n) dst[i] = (__bf16)src[i];
}

// ------------------------------------------------------- GEMM core (async)
// Block 256 thr = 8 waves; tile 128(M) x 64(N); wave does 16x64 (4 tiles).
// Weight tile 64(N) x 32(K) bf16 = 4KB staged in LDS, double-buffered;
// each thread issues one async b128 copy per K-step (256 chunks total).
__device__ __forceinline__ void gemm_tiles_async(
    const __bf16* __restrict__ arow,   // X + (m0+ll)*H
    const __bf16* __restrict__ W,      // [H(N), H(K)] bf16 row-major
    __bf16* smB,                       // [2][64*32] LDS
    int n0, int tid, int lh, int ll, v8f acc[4]) {
  const int row = tid >> 2, ch = tid & 3;   // row 0..63, 16B chunk 0..3
  const __bf16* gsrc = W + (size_t)(n0 + row) * H_ + ch * 8;
  const int ldsoff = row * 32 + ch * 8;

  async_copy16(smB + ldsoff, gsrc);         // prologue: tile k0=0 -> buf0
  for (int k0 = 0; k0 < H_; k0 += 32) {
    const int buf = (k0 >> 5) & 1;
    wait_async0();                          // my copies landed
    __syncthreads();                        // everyone's copies landed
    if (k0 + 32 < H_)
      async_copy16(smB + (buf ^ 1) * (64 * 32) + ldsoff, gsrc + k0 + 32);

    v16bf a = ldfragA(arow + k0 + lh * 16);
    __bf16* bbase = smB + buf * (64 * 32);
#pragma unroll
    for (int t = 0; t < 4; t++)
      acc[t] = wmma_bf16(a, ldfragB(bbase + (t * 16 + ll) * 32 + lh * 16),
                         acc[t]);
  }
}

// ---------------------------------------------------------------- QKV GEMM
// out = X @ W.T + b ;  z=0 -> Q[B,NH,S,HD], z=1 -> K[B,NH,S,HD],
// z=2 -> Vt[B,NH,HD,S] (transposed for contiguous P@V B-fragments).
__global__ __launch_bounds__(256)
void qkv_gemm(const __bf16* __restrict__ X,
              const __bf16* __restrict__ Wqb, const __bf16* __restrict__ Wkb,
              const __bf16* __restrict__ Wvb,
              const float* __restrict__ bq, const float* __restrict__ bk,
              const float* __restrict__ bv,
              __bf16* __restrict__ Q, __bf16* __restrict__ K,
              __bf16* __restrict__ Vt) {
  __shared__ __bf16 smB[2 * 64 * 32];   // 8 KB
  const int ln = threadIdx.x & 31;
  const int w  = threadIdx.x >> 5;
  const int lh = ln >> 4, ll = ln & 15;
  const int m0 = blockIdx.x * 128 + w * 16;
  const int n0 = blockIdx.y * 64;
  const int which = blockIdx.z;

  const __bf16* Wm  = (which == 0) ? Wqb : (which == 1) ? Wkb : Wvb;
  const float* bias = (which == 0) ? bq  : (which == 1) ? bk  : bv;

  v8f zero = {};
  v8f acc[4];
#pragma unroll
  for (int t = 0; t < 4; t++) acc[t] = zero;

  gemm_tiles_async(X + (size_t)(m0 + ll) * H_, Wm, smB, n0,
                   (int)threadIdx.x, lh, ll, acc);

#pragma unroll
  for (int t = 0; t < 4; t++) {
    const int n  = n0 + t * 16 + ll;
    const int h  = n >> 6, hd = n & 63;
    const float bn = bias[n];
#pragma unroll
    for (int r = 0; r < 8; r++) {
      const int m = m0 + r + 8 * lh;
      const int b = m >> 10, s = m & 1023;
      const float val = acc[t][r] + bn;
      if (which == 2)
        Vt[(((size_t)(b * NH_ + h)) * HD_ + hd) * S_ + s] = (__bf16)val;
      else if (which == 0)
        Q[(((size_t)(b * NH_ + h)) * S_ + s) * HD_ + hd] = (__bf16)val;
      else
        K[(((size_t)(b * NH_ + h)) * S_ + s) * HD_ + hd] = (__bf16)val;
    }
  }
}

// ---------------------------------------------------------------- attention
// Flash-style: grid (S/128, B*NH); 8 waves, each owns 16 q-rows x HD=64.
__global__ __launch_bounds__(256)
void attention(const __bf16* __restrict__ Q, const __bf16* __restrict__ K,
               const __bf16* __restrict__ Vt, const float* __restrict__ mask,
               __bf16* __restrict__ ctx) {
  __shared__ __bf16 sm[8 * 16 * 64];     // 2 KB per wave
  const int ln = threadIdx.x & 31, w = threadIdx.x >> 5;
  const int lh = ln >> 4, ll = ln & 15;
  const int bh   = blockIdx.y;           // b*NH + h
  const int bidx = bh >> 4;
  const int hidx = bh & 15;
  const int q0   = blockIdx.x * 128 + w * 16;

  const __bf16* Qb = Q  + (size_t)bh * S_ * HD_;
  const __bf16* Kb = K  + (size_t)bh * S_ * HD_;
  const __bf16* Vb = Vt + (size_t)bh * HD_ * S_;
  const float* mrow = mask + (size_t)bidx * S_;
  __bf16* psm = sm + w * 16 * 64;

  v16bf qf[2];
  const __bf16* qrow = Qb + (size_t)(q0 + ll) * HD_;
#pragma unroll
  for (int s = 0; s < 2; s++) qf[s] = ldfragA(qrow + 32 * s + lh * 16);

  v8f zero = {};
  v8f O[4];
#pragma unroll
  for (int t = 0; t < 4; t++) O[t] = zero;
  float mrun[8], lrun[8];
#pragma unroll
  for (int r = 0; r < 8; r++) { mrun[r] = -__builtin_inff(); lrun[r] = 0.f; }

  for (int kv0 = 0; kv0 < S_; kv0 += 64) {
    // ---- scores = Q @ K^T (4 tiles of 16x16, K-dim 64) ----
    v8f sc[4];
#pragma unroll
    for (int t = 0; t < 4; t++) sc[t] = zero;
#pragma unroll
    for (int t = 0; t < 4; t++) {
#pragma unroll
      for (int s = 0; s < 2; s++) {
        const __bf16* krow =
            Kb + (size_t)(kv0 + t * 16 + ll) * HD_ + 32 * s + lh * 16;
        sc[t] = wmma_bf16(qf[s], ldfragB(krow), sc[t]);
      }
    }
    float mval[4];
#pragma unroll
    for (int t = 0; t < 4; t++) mval[t] = mrow[kv0 + t * 16 + ll];

    // ---- scale + mask + row-max (C layout: lane = kv col, VGPR = q row) ----
    float rmax[8];
#pragma unroll
    for (int r = 0; r < 8; r++) {
      float mx = -__builtin_inff();
#pragma unroll
      for (int t = 0; t < 4; t++) {
        float x = sc[t][r] * 0.125f + mval[t];
        sc[t][r] = x;
        mx = fmaxf(mx, x);
      }
      rmax[r] = mx;
    }
#pragma unroll
    for (int off = 1; off < 16; off <<= 1) {
#pragma unroll
      for (int r = 0; r < 8; r++)
        rmax[r] = fmaxf(rmax[r], __shfl_xor(rmax[r], off, 32));
    }

    // ---- online softmax update ----
    float rsum[8];
#pragma unroll
    for (int r = 0; r < 8; r++) {
      const float mnew  = fmaxf(mrun[r], rmax[r]);
      const float alpha = __expf(mrun[r] - mnew);
      mrun[r] = mnew;
      float sacc = 0.f;
#pragma unroll
      for (int t = 0; t < 4; t++) {
        float p = __expf(sc[t][r] - mnew);
        sc[t][r] = p;
        sacc += p;
      }
      rsum[r] = sacc;
      lrun[r] *= alpha;
#pragma unroll
      for (int t = 0; t < 4; t++) O[t][r] *= alpha;
#pragma unroll
      for (int t = 0; t < 4; t++)
        psm[(r + 8 * lh) * 64 + t * 16 + ll] = (__bf16)sc[t][r];
    }
#pragma unroll
    for (int off = 1; off < 16; off <<= 1) {
#pragma unroll
      for (int r = 0; r < 8; r++) rsum[r] += __shfl_xor(rsum[r], off, 32);
    }
#pragma unroll
    for (int r = 0; r < 8; r++) lrun[r] += rsum[r];

    __syncthreads();

    // ---- O += P @ V : A = probs from LDS, B = Vt rows (contiguous) ----
    v16bf pf[2];
    const __bf16* prow = psm + ll * 64;
#pragma unroll
    for (int s = 0; s < 2; s++) pf[s] = ldfragA(prow + 32 * s + lh * 16);
#pragma unroll
    for (int t = 0; t < 4; t++) {
#pragma unroll
      for (int s = 0; s < 2; s++) {
        const __bf16* vrow =
            Vb + (size_t)(t * 16 + ll) * S_ + kv0 + 32 * s + lh * 16;
        O[t] = wmma_bf16(pf[s], ldfragB(vrow), O[t]);
      }
    }
  }

  // ---- normalize and store ctx as [B,S,H] bf16 ----
#pragma unroll
  for (int r = 0; r < 8; r++) {
    const float inv = 1.f / lrun[r];
    const int q = q0 + r + 8 * lh;
#pragma unroll
    for (int t = 0; t < 4; t++) {
      const int hd = t * 16 + ll;
      ctx[((size_t)(bidx * S_ + q)) * H_ + hidx * HD_ + hd] =
          (__bf16)(O[t][r] * inv);
    }
  }
}

// ---------------------------------------------------------------- out proj
// y = ctx @ Wo.T + bo + hidden   (fp32 result, pre-LayerNorm)
__global__ __launch_bounds__(256)
void proj_gemm(const __bf16* __restrict__ Ctx, const __bf16* __restrict__ Wob,
               const float* __restrict__ bo, const float* __restrict__ hidden,
               float* __restrict__ y) {
  __shared__ __bf16 smB[2 * 64 * 32];   // 8 KB
  const int ln = threadIdx.x & 31;
  const int w  = threadIdx.x >> 5;
  const int lh = ln >> 4, ll = ln & 15;
  const int m0 = blockIdx.x * 128 + w * 16;
  const int n0 = blockIdx.y * 64;

  v8f zero = {};
  v8f acc[4];
#pragma unroll
  for (int t = 0; t < 4; t++) acc[t] = zero;

  gemm_tiles_async(Ctx + (size_t)(m0 + ll) * H_, Wob, smB, n0,
                   (int)threadIdx.x, lh, ll, acc);

#pragma unroll
  for (int t = 0; t < 4; t++) {
    const int n = n0 + t * 16 + ll;
    const float bn = bo[n];
#pragma unroll
    for (int r = 0; r < 8; r++) {
      const int m = m0 + r + 8 * lh;
      y[(size_t)m * H_ + n] = acc[t][r] + bn + hidden[(size_t)m * H_ + n];
    }
  }
}

// ---------------------------------------------------------------- layernorm
__global__ __launch_bounds__(256)
void layernorm(const float* __restrict__ y, const float* __restrict__ gamma,
               const float* __restrict__ beta, float* __restrict__ out) {
  __shared__ float sbuf[256], s2buf[256];
  const int row = blockIdx.x, tid = threadIdx.x;
  const float* p = y + (size_t)row * H_;
  float s = 0.f, s2 = 0.f;
  for (int i = tid; i < H_; i += 256) {
    float v = p[i];
    s += v; s2 += v * v;
  }
  sbuf[tid] = s; s2buf[tid] = s2;
  __syncthreads();
  for (int o = 128; o > 0; o >>= 1) {
    if (tid < o) { sbuf[tid] += sbuf[tid + o]; s2buf[tid] += s2buf[tid + o]; }
    __syncthreads();
  }
  const float mean = sbuf[0] * (1.f / H_);
  const float var  = s2buf[0] * (1.f / H_) - mean * mean;
  const float inv  = rsqrtf(var + 1e-12f);
  for (int i = tid; i < H_; i += 256)
    out[(size_t)row * H_ + i] = (p[i] - mean) * inv * gamma[i] + beta[i];
}

// ---------------------------------------------------------------- launch
extern "C" void kernel_launch(void* const* d_in, const int* in_sizes, int n_in,
                              void* d_out, int out_size, void* d_ws,
                              size_t ws_size, hipStream_t stream) {
  (void)in_sizes; (void)n_in; (void)out_size; (void)ws_size;
  const float* hidden = (const float*)d_in[0];
  const float* mask   = (const float*)d_in[1];
  const float* Wq = (const float*)d_in[2];
  const float* bq = (const float*)d_in[3];
  const float* Wk = (const float*)d_in[4];
  const float* bk = (const float*)d_in[5];
  const float* Wv = (const float*)d_in[6];
  const float* bv = (const float*)d_in[7];
  const float* Wo = (const float*)d_in[8];
  const float* bo = (const float*)d_in[9];
  const float* g  = (const float*)d_in[10];
  const float* bt = (const float*)d_in[11];

  char* ws = (char*)d_ws;
  // [0,16MB)   x_bf16, later reused as ctx_bf16
  // [16,24MB)  Wq/Wk/Wv/Wo bf16 (2MB each)
  // [24,40MB)  Q bf16   |  [24,56MB) reused as y fp32 after attention
  // [40,56MB)  K bf16
  // [56,72MB)  Vt bf16
  __bf16* xb  = (__bf16*)ws;
  __bf16* wqb = (__bf16*)(ws + (16u << 20));
  __bf16* wkb = wqb + (1u << 20);
  __bf16* wvb = wkb + (1u << 20);
  __bf16* wob = wvb + (1u << 20);
  __bf16* Qb  = (__bf16*)(ws + (24u << 20));
  __bf16* Kb  = (__bf16*)(ws + (40u << 20));
  __bf16* Vtb = (__bf16*)(ws + (56u << 20));
  float*  yb  = (float*)(ws + (24u << 20));

  const int nx = M_ * H_;     // 8388608
  const int nw = H_ * H_;     // 1048576
  cvt_f32_bf16<<<nx / 256, 256, 0, stream>>>(hidden, xb, nx);
  cvt_f32_bf16<<<nw / 256, 256, 0, stream>>>(Wq, wqb, nw);
  cvt_f32_bf16<<<nw / 256, 256, 0, stream>>>(Wk, wkb, nw);
  cvt_f32_bf16<<<nw / 256, 256, 0, stream>>>(Wv, wvb, nw);
  cvt_f32_bf16<<<nw / 256, 256, 0, stream>>>(Wo, wob, nw);

  qkv_gemm<<<dim3(M_ / 128, H_ / 64, 3), 256, 0, stream>>>(
      xb, wqb, wkb, wvb, bq, bk, bv, Qb, Kb, Vtb);

  attention<<<dim3(S_ / 128, B_ * NH_), 256, 0, stream>>>(
      Qb, Kb, Vtb, mask, /*ctx=*/xb);

  proj_gemm<<<dim3(M_ / 128, H_ / 64), 256, 0, stream>>>(
      xb, wob, bo, hidden, yb);

  layernorm<<<M_, 256, 0, stream>>>(yb, g, bt, (float*)d_out);
}
